// NonLocalBlock_50706383896595
// MI455X (gfx1250) — matrix operand order repaired and checked by reading, not compiled, not run on hardware
//
#include <hip/hip_runtime.h>
#include <hip/hip_bf16.h>
#include <stdint.h>

// ---------------------------------------------------------------------------
// Problem constants
// ---------------------------------------------------------------------------
#define NB   8
#define CIN  512
#define HWQ  3136      // 56*56 = 196*16
#define CH   128
#define PROJ 384       // 3*CH
static constexpr float ATT_SCALE = 0.08838834764831845f; // 128^-0.5
static constexpr float BN_EPS = 1e-5f;

// ---------------------------------------------------------------------------
// WMMA vector types (CDNA5, wave32)
// ---------------------------------------------------------------------------
typedef __attribute__((ext_vector_type(16))) __bf16 v16bf;
typedef __attribute__((ext_vector_type(8)))  float  v8f;
typedef __attribute__((ext_vector_type(4)))  unsigned int v4u;
typedef __attribute__((ext_vector_type(8)))  int v8i;
typedef __attribute__((ext_vector_type(4)))  int v4i;

union BF16x16 { v16bf v; __bf16 e[16]; uint4 q[2]; };
union F32x8   { v8f   v; float  e[8]; };

__device__ __forceinline__ v8f wmma_bf16(v16bf a, v16bf b, v8f c) {
  // D = A(16x32) * B(32x16) + C(16x16 f32)
  return __builtin_amdgcn_wmma_f32_16x16x32_bf16(false, a, false, b,
                                                 (short)0, c, false, false);
}

// ---------------------------------------------------------------------------
// Tensor Data Mover: 2-D tile load, bf16 elements (data_size = 2B).
// Descriptor layout per CDNA5 ISA ch. 8 (D# groups 0/1; groups 2/3 zero for 2D).
//   tile rows are written to LDS contiguously: LDS[y][x], x fastest.
// ---------------------------------------------------------------------------
__device__ __forceinline__ void tdm_load_2d_bf16(unsigned lds_off, const void* gaddr,
                                                 unsigned tensor_d0, unsigned tensor_d1,
                                                 unsigned tile_d0, unsigned tile_d1,
                                                 unsigned stride0) {
  const unsigned long long ga = (unsigned long long)(uintptr_t)gaddr;
  v4u g0;
  g0[0] = 1u;                                              // count=1, user descriptor
  g0[1] = lds_off;                                         // lds_addr [63:32]
  g0[2] = (unsigned)(ga & 0xFFFFFFFFu);                    // global_addr [95:64]
  g0[3] = (unsigned)((ga >> 32) & 0x01FFFFFFu) | (2u << 30); // addr hi | type=2
  v8i g1;
  g1[0] = (int)(1u << 16);                                 // wg_mask=0, data_size=1 (2B)
  g1[1] = (int)((tensor_d0 & 0xFFFFu) << 16);              // tensor_dim0 [79:48] lo
  g1[2] = (int)((tensor_d0 >> 16) | ((tensor_d1 & 0xFFFFu) << 16)); // td0 hi | td1 lo
  g1[3] = (int)((tensor_d1 >> 16) | ((tile_d0 & 0xFFFFu) << 16));   // td1 hi | tile_dim0
  g1[4] = (int)(tile_d1 & 0xFFFFu);                        // tile_dim1 | tile_dim2=0
  g1[5] = (int)stride0;                                    // tensor_dim0_stride lo32
  g1[6] = 0;                                               // stride0 hi | stride1 lo
  g1[7] = 0;                                               // stride1 hi
  v4i z4 = {0, 0, 0, 0};
#if defined(__clang_major__) && (__clang_major__ >= 23)
  v8i z8 = {0, 0, 0, 0, 0, 0, 0, 0};
  __builtin_amdgcn_tensor_load_to_lds(g0, g1, z4, z4, z8, 0);
#else
  __builtin_amdgcn_tensor_load_to_lds(g0, g1, z4, z4, 0);
#endif
}

// ---------------------------------------------------------------------------
// Kernel 0: prep — bf16 weight conversion + folded BN coefficients
// ---------------------------------------------------------------------------
__global__ void prep_kernel(const float* __restrict__ W_w,
                            const float* __restrict__ Z_w,
                            const float* __restrict__ gamma,
                            const float* __restrict__ beta,
                            const float* __restrict__ mean,
                            const float* __restrict__ var,
                            __bf16* __restrict__ Wb,
                            __bf16* __restrict__ Zb,
                            float* __restrict__ inv,
                            float* __restrict__ shift) {
  int i = blockIdx.x * blockDim.x + threadIdx.x;
  if (i < PROJ * CIN) Wb[i] = (__bf16)W_w[i];
  if (i < CIN * CH)   Zb[i] = (__bf16)Z_w[i];
  if (i < CIN) {
    float iv = gamma[i] * rsqrtf(var[i] + BN_EPS);
    inv[i]   = iv;
    shift[i] = beta[i] - mean[i] * iv;
  }
}

// ---------------------------------------------------------------------------
// Kernel 1: proj GEMM  proj[o,q] = sum_c W[o,c] * x[n,c,q] + bias[o]
//   o<128  -> Q[n][q][c]   (transposed store, c-contiguous)
//   o<256  -> K[n][q][c]
//   else   -> Vt[n][c][q]  (natural store, q-contiguous)
// One wave per 16x16 output tile, K-loop over 512 in steps of 32.
// ---------------------------------------------------------------------------
__global__ __launch_bounds__(32)
void proj_kernel(const float* __restrict__ x,
                 const __bf16* __restrict__ Wb,
                 const float* __restrict__ Wbias,
                 __bf16* __restrict__ Q,
                 __bf16* __restrict__ Kd,
                 __bf16* __restrict__ Vt) {
  __shared__ float xs[32][17];          // x tile [c][q], padded vs bank conflicts

  const int lane = threadIdx.x;
  const int half = lane >> 4;
  const int l15  = lane & 15;
  const int q0 = blockIdx.x * 16;
  const int o0 = blockIdx.y * 16;
  const int n  = blockIdx.z;
  const float* xn = x + (size_t)n * CIN * HWQ;

  F32x8 acc;
#pragma unroll
  for (int r = 0; r < 8; ++r) acc.e[r] = 0.0f;

  for (int c0 = 0; c0 < CIN; c0 += 32) {
    // --- stage x tile [32c][16q] coalesced: lane loads one row of 16 floats
    {
      const float* src = xn + (size_t)(c0 + lane) * HWQ + q0;
      float4 a = *(const float4*)(src + 0);
      float4 b = *(const float4*)(src + 4);
      float4 c = *(const float4*)(src + 8);
      float4 d = *(const float4*)(src + 12);
      xs[lane][0] = a.x;  xs[lane][1] = a.y;  xs[lane][2]  = a.z;  xs[lane][3]  = a.w;
      xs[lane][4] = b.x;  xs[lane][5] = b.y;  xs[lane][6]  = b.z;  xs[lane][7]  = b.w;
      xs[lane][8] = c.x;  xs[lane][9] = c.y;  xs[lane][10] = c.z;  xs[lane][11] = c.w;
      xs[lane][12] = d.x; xs[lane][13] = d.y; xs[lane][14] = d.z;  xs[lane][15] = d.w;
      if (c0 + 32 < CIN)
        __builtin_prefetch(src + (size_t)32 * HWQ, 0, 1);   // global_prefetch
    }
    __syncthreads();

    // --- A fragment: W tile, M=o rows; lane row = o0+l15
    //     lanes<16: K = 0..15 ; lanes>=16: K = 16..31
    BF16x16 afr;
    const __bf16* wrow = Wb + (size_t)(o0 + l15) * CIN + c0;
    afr.q[0] = *(const uint4*)(wrow + 16 * half);
    afr.q[1] = *(const uint4*)(wrow + 16 * half + 8);

    // --- B fragment: x tile (Kdim=c, N=q); lane col q=l15, c = 16*half + 2v,2v+1
    BF16x16 bfr;
#pragma unroll
    for (int v = 0; v < 8; ++v) {
      bfr.e[2 * v]     = (__bf16)xs[16 * half + 2 * v][l15];
      bfr.e[2 * v + 1] = (__bf16)xs[16 * half + 2 * v + 1][l15];
    }
    __syncthreads();

    acc.v = wmma_bf16(afr.v, bfr.v, acc.v);
  }

  // epilogue: bias + scatter into Q/K/Vt (bf16)
#pragma unroll
  for (int r = 0; r < 8; ++r) {
    const int o = o0 + r + 8 * half;
    const int q = q0 + l15;
    const float val = acc.e[r] + Wbias[o];
    const __bf16 bv = (__bf16)val;
    if (o < CH)
      Q[((size_t)n * HWQ + q) * CH + o] = bv;
    else if (o < 2 * CH)
      Kd[((size_t)n * HWQ + q) * CH + (o - CH)] = bv;
    else
      Vt[((size_t)n * CH + (o - 2 * CH)) * HWQ + q] = bv;
  }
}

// ---------------------------------------------------------------------------
// Kernel 2: fused flash-attention
//   per wave: 16 q-rows; loop k in steps of 32; O (16x128 f32) in VGPRs.
//   K/V tiles double-buffered in LDS, staged by the Tensor Data Mover:
//   wave 0 issues tensor_load_to_lds for tile i+1 while all waves compute
//   tile i, then s_wait_tensorcnt(2) + barrier. TDM completions are in-order.
// ---------------------------------------------------------------------------
__global__ __launch_bounds__(128)
void attn_kernel(const __bf16* __restrict__ Q,
                 const __bf16* __restrict__ Kd,
                 const __bf16* __restrict__ Vt,
                 __bf16* __restrict__ att) {
  __shared__ __bf16 Kt[2][32][CH];     // 16 KB  [buf][k_local][c]
  __shared__ __bf16 Vts[2][CH][32];    // 16 KB  [buf][c][k_local]
  __shared__ __bf16 Pld[4][16][32];    //  4 KB  per-wave P tile

  const int tid  = threadIdx.x;
  const int wave = tid >> 5;
  const int lane = tid & 31;
  const int half = lane >> 4;
  const int l15  = lane & 15;
  const int n  = blockIdx.y;
  const int q0 = (blockIdx.x * 4 + wave) * 16;

  const __bf16* Qn = Q  + (size_t)n * HWQ * CH;
  const __bf16* Kn = Kd + (size_t)n * HWQ * CH;
  const __bf16* Vn = Vt + (size_t)n * CH * HWQ;

  const unsigned ktOff[2] = { (unsigned)(uintptr_t)&Kt[0][0][0],
                              (unsigned)(uintptr_t)&Kt[1][0][0] };
  const unsigned vtOff[2] = { (unsigned)(uintptr_t)&Vts[0][0][0],
                              (unsigned)(uintptr_t)&Vts[1][0][0] };

  // --- Q A-fragments for the four 32-wide c chunks (persist across k loop)
  BF16x16 qa[4];
  {
    const __bf16* qrow = Qn + (size_t)(q0 + l15) * CH;
#pragma unroll
    for (int cc = 0; cc < 4; ++cc) {
      qa[cc].q[0] = *(const uint4*)(qrow + cc * 32 + 16 * half);
      qa[cc].q[1] = *(const uint4*)(qrow + cc * 32 + 16 * half + 8);
    }
  }

  float m[8], lsum[8];
  F32x8 acc[8];
#pragma unroll
  for (int r = 0; r < 8; ++r) { m[r] = -1e30f; lsum[r] = 0.0f; }
#pragma unroll
  for (int j = 0; j < 8; ++j)
#pragma unroll
    for (int r = 0; r < 8; ++r) acc[j].e[r] = 0.0f;

  const int NIT = HWQ / 32;   // 98 k-tiles

  // prologue: TDM-stage tile 0 into buffer 0
  if (wave == 0) {
    // K tile: 32 rows (k) x 128 cols (c) from [HW][CH]
    tdm_load_2d_bf16(ktOff[0], Kn, CH, HWQ, CH, 32, CH);
    // V tile: 128 rows (c) x 32 cols (k) from [CH][HW]
    tdm_load_2d_bf16(vtOff[0], Vn, HWQ, CH, 32, CH, HWQ);
  }

  for (int it = 0; it < NIT; ++it) {
    const int cur = it & 1;

    if (wave == 0) {
      if (it + 1 < NIT) {
        const int k1 = (it + 1) * 32;
        tdm_load_2d_bf16(ktOff[cur ^ 1], Kn + (size_t)k1 * CH, CH, HWQ, CH, 32, CH);
        tdm_load_2d_bf16(vtOff[cur ^ 1], Vn + k1, HWQ, CH, 32, CH, HWQ);
        // 2 just-issued ops may stay in flight; current tile's 2 are done.
        __builtin_amdgcn_s_wait_tensorcnt(2);
      } else {
        __builtin_amdgcn_s_wait_tensorcnt(0);
      }
    }
    __syncthreads();

    // --- S tiles: s0 = Q . K[k0..k0+15]^T , s1 = Q . K[k0+16..k0+31]^T
    F32x8 S0, S1;
#pragma unroll
    for (int r = 0; r < 8; ++r) { S0.e[r] = 0.0f; S1.e[r] = 0.0f; }
#pragma unroll
    for (int cc = 0; cc < 4; ++cc) {
      BF16x16 kb0, kb1;
      kb0.v = *(const v16bf*)(&Kt[cur][l15][cc * 32 + 16 * half]);
      kb1.v = *(const v16bf*)(&Kt[cur][16 + l15][cc * 32 + 16 * half]);
      S0.v = wmma_bf16(qa[cc].v, kb0.v, S0.v);
      S1.v = wmma_bf16(qa[cc].v, kb1.v, S1.v);
    }

    // --- online softmax update (row q = r + 8*half, k across 16 lanes)
#pragma unroll
    for (int r = 0; r < 8; ++r) {
      float a = S0.e[r] * ATT_SCALE;
      float b = S1.e[r] * ATT_SCALE;
      S0.e[r] = a; S1.e[r] = b;
      float mx = fmaxf(a, b);
#pragma unroll
      for (int d = 8; d >= 1; d >>= 1) mx = fmaxf(mx, __shfl_xor(mx, d, 32));
      const float mnew = fmaxf(m[r], mx);
      const float p0 = __expf(S0.e[r] - mnew);
      const float p1 = __expf(S1.e[r] - mnew);
      S0.e[r] = p0; S1.e[r] = p1;
      float rsum = p0 + p1;
#pragma unroll
      for (int d = 8; d >= 1; d >>= 1) rsum += __shfl_xor(rsum, d, 32);
      const float alpha = __expf(m[r] - mnew);
      lsum[r] = lsum[r] * alpha + rsum;
      m[r] = mnew;
#pragma unroll
      for (int j = 0; j < 8; ++j) acc[j].e[r] *= alpha;
    }

    // --- P: C-layout -> A-layout via per-wave LDS bounce
#pragma unroll
    for (int r = 0; r < 8; ++r) {
      Pld[wave][r + 8 * half][l15]      = (__bf16)S0.e[r];
      Pld[wave][r + 8 * half][16 + l15] = (__bf16)S1.e[r];
    }
    BF16x16 pa;
    pa.q[0] = *(const uint4*)(&Pld[wave][l15][16 * half]);
    pa.q[1] = *(const uint4*)(&Pld[wave][l15][16 * half + 8]);

    // --- O += P(16x32) . V(32x128)
#pragma unroll
    for (int j = 0; j < 8; ++j) {
      BF16x16 vb;
      vb.v = *(const v16bf*)(&Vts[cur][j * 16 + l15][16 * half]);
      acc[j].v = wmma_bf16(pa.v, vb.v, acc[j].v);
    }
    __syncthreads();   // all waves done with buf `cur` before TDM reuses it
  }

  // --- normalize and store att[n][q][c] (bf16, c-contiguous)
#pragma unroll
  for (int j = 0; j < 8; ++j) {
#pragma unroll
    for (int r = 0; r < 8; ++r) {
      const int q = q0 + r + 8 * half;
      const int c = j * 16 + l15;
      att[((size_t)n * HWQ + q) * CH + c] = (__bf16)(acc[j].e[r] / lsum[r]);
    }
  }
}

// ---------------------------------------------------------------------------
// Kernel 3: z = Z_w @ att^T, fused BN affine + residual, fp32 out
// One wave per 16(o) x 16(q) tile, K-loop over CH=128.
// ---------------------------------------------------------------------------
__global__ __launch_bounds__(32)
void out_kernel(const __bf16* __restrict__ att,
                const __bf16* __restrict__ Zb,
                const float* __restrict__ inv,
                const float* __restrict__ shift,
                const float* __restrict__ x,
                float* __restrict__ out) {
  const int lane = threadIdx.x;
  const int half = lane >> 4;
  const int l15  = lane & 15;
  const int q0 = blockIdx.x * 16;
  const int o0 = blockIdx.y * 16;
  const int n  = blockIdx.z;

  F32x8 acc;
#pragma unroll
  for (int r = 0; r < 8; ++r) acc.e[r] = 0.0f;

#pragma unroll
  for (int c0 = 0; c0 < CH; c0 += 32) {
    BF16x16 a, b;
    const __bf16* zrow = Zb + (size_t)(o0 + l15) * CH + c0;
    a.q[0] = *(const uint4*)(zrow + 16 * half);
    a.q[1] = *(const uint4*)(zrow + 16 * half + 8);
    b.v = *(const v16bf*)(att + ((size_t)n * HWQ + q0 + l15) * CH + c0 + 16 * half);
    acc.v = wmma_bf16(a.v, b.v, acc.v);
  }

#pragma unroll
  for (int r = 0; r < 8; ++r) {
    const int o = o0 + r + 8 * half;
    const int q = q0 + l15;
    const size_t idx = ((size_t)n * CIN + o) * HWQ + q;
    out[idx] = acc.e[r] * inv[o] + shift[o] + x[idx];
  }
}

// ---------------------------------------------------------------------------
// Launch
// ---------------------------------------------------------------------------
extern "C" void kernel_launch(void* const* d_in, const int* in_sizes, int n_in,
                              void* d_out, int out_size, void* d_ws, size_t ws_size,
                              hipStream_t stream) {
  const float* x     = (const float*)d_in[0];
  const float* W_w   = (const float*)d_in[1];
  const float* W_b   = (const float*)d_in[2];
  const float* Z_w   = (const float*)d_in[3];
  const float* gamma = (const float*)d_in[4];
  const float* beta  = (const float*)d_in[5];
  const float* mean  = (const float*)d_in[6];
  const float* var   = (const float*)d_in[7];
  float* out = (float*)d_out;

  char* ws = (char*)d_ws;
  // workspace layout (all 256-byte aligned)
  __bf16* Wb    = (__bf16*)(ws + 0);                    //  384*512 bf16
  __bf16* Zb    = (__bf16*)(ws + 393216);               //  512*128 bf16
  float*  inv   = (float*)(ws + 524288);                //  512 f32
  float*  shift = (float*)(ws + 526336);                //  512 f32
  __bf16* Q     = (__bf16*)(ws + 528384);               //  N*HW*CH bf16
  __bf16* Kd    = (__bf16*)(ws + 6950912);
  __bf16* Vt    = (__bf16*)(ws + 13373440);
  __bf16* att   = (__bf16*)(ws + 19795968);             // ends at 26218496 B

  prep_kernel<<<(PROJ * CIN + 255) / 256, 256, 0, stream>>>(
      W_w, Z_w, gamma, beta, mean, var, Wb, Zb, inv, shift);

  dim3 g1(HWQ / 16, PROJ / 16, NB);
  proj_kernel<<<g1, 32, 0, stream>>>(x, Wb, W_b, Q, Kd, Vt);

  dim3 g2(HWQ / 64, NB);
  attn_kernel<<<g2, 128, 0, stream>>>(Q, Kd, Vt, att);

  dim3 g3(HWQ / 16, CIN / 16, NB);
  out_kernel<<<g3, 32, 0, stream>>>(att, Zb, inv, shift, x, out);
}
